// GATv2_18073222382226
// MI455X (gfx1250) — matrix operand order repaired and checked by reading, not compiled, not run on hardware
//
#include <hip/hip_runtime.h>
#include <hip/hip_bf16.h>

// ---------------------------------------------------------------------------
// GATv2 (2 conv layers) + GraphNorm + MLP for MI455X (gfx1250, wave32, WMMA)
// Dense projections: v_wmma_f32_16x16x32_f16, 4x M-blocked (64x16 per wave)
// so each B fragment feeds 4 WMMAs. Edge phase: wave32-coalesced gather /
// segment-softmax (ordered-uint atomicMax) / atomic scatter.
// ---------------------------------------------------------------------------

#define NN   50000
#define EE   800000
#define ET   850000      // EE + NN self loops
#define INF_ 128
#define HH   4
#define CC   64
#define HC   256         // HH*CC
#define HID  64
#define OUTD 2
#define MBLK 4           // M-subtiles (16 rows each) per wave

typedef __attribute__((ext_vector_type(16))) _Float16 v16h;
typedef __attribute__((ext_vector_type(8)))  float    v8f;

// -------- order-preserving float <-> uint key for atomicMax on floats -------
__device__ __forceinline__ unsigned fkey(float f) {
  unsigned u = __float_as_uint(f);
  return (u & 0x80000000u) ? ~u : (u | 0x80000000u);
}
__device__ __forceinline__ float keyf(unsigned k) {
  unsigned u = (k & 0x80000000u) ? (k & 0x7fffffffu) : ~k;
  return __uint_as_float(u);
}
#define NEG_INF_KEY 0x007FFFFFu   // fkey(-inf)

// ---------------------------------------------------------------------------
// WMMA GEMM: Y[M,Nout] = X[M,K] @ W[K,Nout] + bias (optional ReLU).
// One wave computes a 64x16 output block (4 stacked 16x16 WMMA tiles sharing
// one B fragment per K-step). Nout%16==0, K%32==0; M tail handled by
// clamped A loads + guarded stores (wave-uniform control; EXEC all-ones).
// Fragment layouts used here (contiguous k, same for A and B):
//   A: row M = lane%16, K = 16*(lane/16) + p.
//   B: N = lane%16, K = 16*(lane/16) + p.
//   C: M = r + 8*(lane/16), N = lane%16.
// ---------------------------------------------------------------------------
__global__ __launch_bounds__(256) void wmma_gemm_kernel(
    const float* __restrict__ X, const float* __restrict__ W,
    const float* __restrict__ bias, float* __restrict__ Y,
    int M, int K, int Nout, int do_relu)
{
  const int wavesPerBlock = blockDim.x >> 5;
  const int waveId = blockIdx.x * wavesPerBlock + (threadIdx.x >> 5);
  const int lane   = threadIdx.x & 31;
  const int tilesN = Nout >> 4;
  const int mGroups = (M + 63) >> 6;              // 64-row groups
  if (waveId >= mGroups * tilesN) return;         // wave-uniform exit

  const int tileM = (waveId / tilesN) << 6;
  const int tileN = (waveId % tilesN) << 4;
  const int hs = lane >> 4;                       // lane-half select
  const int lr = lane & 15;

  v8f acc[MBLK] = {};
  const float* xrow[MBLK];
#pragma unroll
  for (int t = 0; t < MBLK; ++t) {
    int r = tileM + 16 * t + lr;
    if (r >= M) r = M - 1;                        // clamp: loads stay in-bounds
    xrow[t] = X + (size_t)r * K;
  }

  for (int kb = 0; kb < K; kb += 32) {
    // B fragment (shared by all 4 M-subtiles): K = kb + 16*hs + p
    v16h b;
    const float* wcol = W + (size_t)(kb + (hs << 4)) * Nout + (tileN + lr);
#pragma unroll
    for (int p = 0; p < 16; ++p) b[p] = (_Float16)wcol[(size_t)p * Nout];

#pragma unroll
    for (int t = 0; t < MBLK; ++t) {
      // A fragment: two contiguous 8-float runs per lane-half, as 4x b128.
      const float4* alo = (const float4*)(xrow[t] + kb + (hs << 4));
      const float4* ahi = (const float4*)(xrow[t] + kb + 8 + (hs << 4));
      float4 l0 = alo[0], l1 = alo[1], h0 = ahi[0], h1 = ahi[1];
      v16h a;
      a[0]  = (_Float16)l0.x; a[1]  = (_Float16)l0.y;
      a[2]  = (_Float16)l0.z; a[3]  = (_Float16)l0.w;
      a[4]  = (_Float16)l1.x; a[5]  = (_Float16)l1.y;
      a[6]  = (_Float16)l1.z; a[7]  = (_Float16)l1.w;
      a[8]  = (_Float16)h0.x; a[9]  = (_Float16)h0.y;
      a[10] = (_Float16)h0.z; a[11] = (_Float16)h0.w;
      a[12] = (_Float16)h1.x; a[13] = (_Float16)h1.y;
      a[14] = (_Float16)h1.z; a[15] = (_Float16)h1.w;
      acc[t] = __builtin_amdgcn_wmma_f32_16x16x32_f16(
          false, a, false, b, (short)0, acc[t], false, false);
    }
  }

  const int col = tileN + lr;
  const float bv = bias ? bias[col] : 0.0f;
#pragma unroll
  for (int t = 0; t < MBLK; ++t) {
#pragma unroll
    for (int r = 0; r < 8; ++r) {
      int row = tileM + 16 * t + r + (hs ? 8 : 0);
      if (row < M) {
        float v = acc[t][r] + bv;
        if (do_relu) v = fmaxf(v, 0.0f);
        Y[(size_t)row * Nout + col] = v;
      }
    }
  }
}

// ---------------------------------------------------------------------------
// Per-conv init: m keys <- -inf key, s <- 0, out[i][f] <- bias[f]
// ---------------------------------------------------------------------------
__global__ void conv_init_kernel(unsigned* __restrict__ mkey,
                                 float* __restrict__ sbuf,
                                 float* __restrict__ outbuf,
                                 const float* __restrict__ bias)
{
  int idx = blockIdx.x * blockDim.x + threadIdx.x;
  if (idx < NN * HH) { mkey[idx] = NEG_INF_KEY; sbuf[idx] = 0.0f; }
  if (idx < NN * HC) outbuf[idx] = bias[idx & (HC - 1)];
}

// ---------------------------------------------------------------------------
// Edge scores: one wave per (edge, head).
// e = sum_c att[h,c] * leaky_relu(xl[src,h,c] + xr[dst,h,c], 0.2)
// atomicMax segment max over dst (ordered-uint keys).
// ---------------------------------------------------------------------------
__global__ __launch_bounds__(256) void edge_score_kernel(
    const float* __restrict__ xl, const float* __restrict__ xr,
    const float* __restrict__ att,
    const int* __restrict__ srcA, const int* __restrict__ dstA,
    float* __restrict__ ebuf, unsigned* __restrict__ mkey)
{
  const int wid = blockIdx.x * (blockDim.x >> 5) + (threadIdx.x >> 5);
  if (wid >= ET * HH) return;
  const int lane = threadIdx.x & 31;
  const int eidx = wid >> 2;
  const int h    = wid & 3;
  const int s = (eidx < EE) ? srcA[eidx] : (eidx - EE);
  const int d = (eidx < EE) ? dstA[eidx] : (eidx - EE);

  const float* pl = xl + (size_t)s * HC + h * CC;
  const float* pr = xr + (size_t)d * HC + h * CC;
  const float* pa = att + h * CC;

  float sum = 0.0f;
#pragma unroll
  for (int c0 = 0; c0 < CC; c0 += 32) {
    int c = c0 + lane;
    float v = pl[c] + pr[c];
    v = (v > 0.0f) ? v : 0.2f * v;     // leaky_relu(0.2)
    sum += pa[c] * v;
  }
#pragma unroll
  for (int off = 16; off > 0; off >>= 1) sum += __shfl_down(sum, off, 32);

  if (lane == 0) {
    ebuf[wid] = sum;
    atomicMax(&mkey[d * HH + h], fkey(sum));
  }
}

// ---------------------------------------------------------------------------
// ex = exp(e - m[dst]); segment sum over dst. One thread per (edge, head).
// ---------------------------------------------------------------------------
__global__ void edge_exp_kernel(const int* __restrict__ srcA,
                                const int* __restrict__ dstA,
                                float* __restrict__ ebuf,
                                const unsigned* __restrict__ mkey,
                                float* __restrict__ sbuf)
{
  int idx = blockIdx.x * blockDim.x + threadIdx.x;
  if (idx >= ET * HH) return;
  int eidx = idx >> 2, h = idx & 3;
  int d = (eidx < EE) ? dstA[eidx] : (eidx - EE);
  float m = keyf(mkey[d * HH + h]);
  float ex = __expf(ebuf[idx] - m);
  ebuf[idx] = ex;
  atomicAdd(&sbuf[d * HH + h], ex);
}

// ---------------------------------------------------------------------------
// Scatter: out[dst, h*64+c] += alpha * xl[src, h*64+c]. One wave per (e,h);
// lanes cover c and c+32 -> contiguous 128B atomic bursts.
// ---------------------------------------------------------------------------
__global__ __launch_bounds__(256) void edge_scatter_kernel(
    const float* __restrict__ xl, const float* __restrict__ ebuf,
    const float* __restrict__ sbuf,
    const int* __restrict__ srcA, const int* __restrict__ dstA,
    float* __restrict__ outbuf)
{
  const int wid = blockIdx.x * (blockDim.x >> 5) + (threadIdx.x >> 5);
  if (wid >= ET * HH) return;
  const int lane = threadIdx.x & 31;
  const int eidx = wid >> 2, h = wid & 3;
  const int s = (eidx < EE) ? srcA[eidx] : (eidx - EE);
  const int d = (eidx < EE) ? dstA[eidx] : (eidx - EE);

  const float alpha = ebuf[wid] / (sbuf[d * HH + h] + 1e-16f);
  const float* pl = xl + (size_t)s * HC + h * CC;
  float* po = outbuf + (size_t)d * HC + h * CC;
  atomicAdd(&po[lane],      alpha * pl[lane]);
  atomicAdd(&po[lane + 32], alpha * pl[lane + 32]);
}

// ---------------------------------------------------------------------------
// GraphNorm (single graph): column stats over all nodes, then normalize+ReLU.
// ---------------------------------------------------------------------------
__global__ void zero_kernel(float* __restrict__ p, int n) {
  int i = blockIdx.x * blockDim.x + threadIdx.x;
  if (i < n) p[i] = 0.0f;
}

__global__ __launch_bounds__(256) void gn_sum_kernel(
    const float* __restrict__ h, float* __restrict__ sums, int rowsPerBlock)
{
  int f  = threadIdx.x;                    // 0..255 == column
  int r0 = blockIdx.x * rowsPerBlock;
  int r1 = r0 + rowsPerBlock; if (r1 > NN) r1 = NN;
  float acc = 0.0f;
  for (int r = r0; r < r1; ++r) acc += h[(size_t)r * HC + f];
  atomicAdd(&sums[f], acc);
}

__global__ __launch_bounds__(256) void gn_var_kernel(
    const float* __restrict__ h, const float* __restrict__ sums,
    const float* __restrict__ ms, float* __restrict__ vars, int rowsPerBlock)
{
  int f  = threadIdx.x;
  float m = ms[f] * (sums[f] / (float)NN);
  int r0 = blockIdx.x * rowsPerBlock;
  int r1 = r0 + rowsPerBlock; if (r1 > NN) r1 = NN;
  float acc = 0.0f;
  for (int r = r0; r < r1; ++r) { float xc = h[(size_t)r * HC + f] - m; acc += xc * xc; }
  atomicAdd(&vars[f], acc);
}

__global__ void gn_norm_relu_kernel(float* __restrict__ h,
                                    const float* __restrict__ sums,
                                    const float* __restrict__ vars,
                                    const float* __restrict__ ms,
                                    const float* __restrict__ w,
                                    const float* __restrict__ b)
{
  int idx = blockIdx.x * blockDim.x + threadIdx.x;
  if (idx >= NN * HC) return;
  int f = idx & (HC - 1);
  float mean = sums[f] / (float)NN;
  float var  = vars[f] / (float)NN;
  float xc   = h[idx] - ms[f] * mean;
  float y    = w[f] * xc * rsqrtf(var + 1e-5f) + b[f];
  h[idx] = fmaxf(y, 0.0f);
}

// ---------------------------------------------------------------------------
// Final tiny projection: out[N,2] = h[N,64] @ w[64,2] + b
// ---------------------------------------------------------------------------
__global__ void lin2_kernel(const float* __restrict__ h,
                            const float* __restrict__ w,
                            const float* __restrict__ b,
                            float* __restrict__ out)
{
  int idx = blockIdx.x * blockDim.x + threadIdx.x;
  if (idx >= NN * OUTD) return;
  int i = idx >> 1, o = idx & 1;
  float acc = b[o];
  const float* hr = h + (size_t)i * HID;
#pragma unroll
  for (int k = 0; k < HID; ++k) acc += hr[k] * w[k * OUTD + o];
  out[idx] = acc;
}

// ---------------------------------------------------------------------------
// Host-side helpers
// ---------------------------------------------------------------------------
static inline void launch_gemm(const float* X, const float* W, const float* bias,
                               float* Y, int M, int K, int Nout, int relu,
                               hipStream_t s)
{
  int mGroups = (M + 63) >> 6;
  int totalWaves = mGroups * (Nout >> 4);
  int blocks = (totalWaves + 7) / 8;         // 8 waves / 256-thread block
  wmma_gemm_kernel<<<blocks, 256, 0, s>>>(X, W, bias, Y, M, K, Nout, relu);
}

static void run_gat_conv(const float* xin, int K,
                         const float* wl, const float* bl,
                         const float* wr, const float* br,
                         const float* att, const float* bias,
                         const int* srcA, const int* dstA,
                         float* xl, float* xr, float* ebuf,
                         unsigned* mkey, float* sbuf, float* outbuf,
                         hipStream_t s)
{
  launch_gemm(xin, wl, bl, xl, NN, K, HC, 0, s);
  launch_gemm(xin, wr, br, xr, NN, K, HC, 0, s);

  conv_init_kernel<<<(NN * HC + 255) / 256, 256, 0, s>>>(mkey, sbuf, outbuf, bias);

  int edgeWaves  = ET * HH;
  int edgeBlocks = (edgeWaves + 7) / 8;
  edge_score_kernel<<<edgeBlocks, 256, 0, s>>>(xl, xr, att, srcA, dstA, ebuf, mkey);
  edge_exp_kernel<<<(edgeWaves + 255) / 256, 256, 0, s>>>(srcA, dstA, ebuf, mkey, sbuf);
  edge_scatter_kernel<<<edgeBlocks, 256, 0, s>>>(xl, ebuf, sbuf, srcA, dstA, outbuf);
}

static void run_graph_norm(float* h, const float* w, const float* b,
                           const float* ms, float* sums, float* vars,
                           hipStream_t s)
{
  zero_kernel<<<2, 256, 0, s>>>(sums, HC);
  zero_kernel<<<2, 256, 0, s>>>(vars, HC);
  const int rowsPerBlock = 64;
  const int blocks = (NN + rowsPerBlock - 1) / rowsPerBlock;
  gn_sum_kernel<<<blocks, 256, 0, s>>>(h, sums, rowsPerBlock);
  gn_var_kernel<<<blocks, 256, 0, s>>>(h, sums, ms, vars, rowsPerBlock);
  gn_norm_relu_kernel<<<(NN * HC + 255) / 256, 256, 0, s>>>(h, sums, vars, ms, w, b);
}

extern "C" void kernel_launch(void* const* d_in, const int* in_sizes, int n_in,
                              void* d_out, int out_size, void* d_ws, size_t ws_size,
                              hipStream_t stream)
{
  // ---- inputs (setup_inputs dict order) ----
  const float* x        = (const float*)d_in[0];
  const int*   ei       = (const int*)  d_in[1];
  const float* c0_wl    = (const float*)d_in[2];
  const float* c0_bl    = (const float*)d_in[3];
  const float* c0_wr    = (const float*)d_in[4];
  const float* c0_br    = (const float*)d_in[5];
  const float* c0_att   = (const float*)d_in[6];
  const float* c0_bias  = (const float*)d_in[7];
  const float* c1_wl    = (const float*)d_in[8];
  const float* c1_bl    = (const float*)d_in[9];
  const float* c1_wr    = (const float*)d_in[10];
  const float* c1_br    = (const float*)d_in[11];
  const float* c1_att   = (const float*)d_in[12];
  const float* c1_bias  = (const float*)d_in[13];
  const float* gn0_w    = (const float*)d_in[14];
  const float* gn0_b    = (const float*)d_in[15];
  const float* gn0_ms   = (const float*)d_in[16];
  const float* gn1_w    = (const float*)d_in[17];
  const float* gn1_b    = (const float*)d_in[18];
  const float* gn1_ms   = (const float*)d_in[19];
  const float* lin0_w   = (const float*)d_in[20];
  const float* lin0_b   = (const float*)d_in[21];
  const float* lin1_w   = (const float*)d_in[22];
  const float* lin1_b   = (const float*)d_in[23];
  const float* lin2_w   = (const float*)d_in[24];
  const float* lin2_b   = (const float*)d_in[25];
  float* out = (float*)d_out;

  const int* srcA = ei;            // edge_index[0]
  const int* dstA = ei + EE;       // edge_index[1]

  // ---- workspace layout (fp32) ----
  float* ws   = (float*)d_ws;
  float* xl   = ws;                       // NN*HC
  float* xr   = xl   + (size_t)NN * HC;   // NN*HC
  float* hbuf = xr   + (size_t)NN * HC;   // NN*HC
  float* ebuf = hbuf + (size_t)NN * HC;   // ET*HH
  float* mbuf = ebuf + (size_t)ET * HH;   // NN*HH  (uint keys)
  float* sbuf = mbuf + (size_t)NN * HH;   // NN*HH
  float* sums = sbuf + (size_t)NN * HH;   // HC
  float* vars = sums + HC;                // HC
  float* hid0 = vars + HC;                // NN*HID
  float* hid1 = hid0 + (size_t)NN * HID;  // NN*HID
  unsigned* mkey = (unsigned*)mbuf;

  // ---- conv0: x[N,128] -> hbuf[N,256] ----
  run_gat_conv(x, INF_, c0_wl, c0_bl, c0_wr, c0_br, c0_att, c0_bias,
               srcA, dstA, xl, xr, ebuf, mkey, sbuf, hbuf, stream);
  run_graph_norm(hbuf, gn0_w, gn0_b, gn0_ms, sums, vars, stream);

  // ---- conv1: hbuf[N,256] -> hbuf[N,256] (xl/xr computed first, then reuse) ----
  run_gat_conv(hbuf, HC, c1_wl, c1_bl, c1_wr, c1_br, c1_att, c1_bias,
               srcA, dstA, xl, xr, ebuf, mkey, sbuf, hbuf, stream);
  run_graph_norm(hbuf, gn1_w, gn1_b, gn1_ms, sums, vars, stream);

  // ---- MLP head ----
  launch_gemm(hbuf, lin0_w, lin0_b, hid0, NN, HC,  HID, 1, stream);  // relu
  launch_gemm(hid0, lin1_w, lin1_b, hid1, NN, HID, HID, 1, stream);  // relu
  lin2_kernel<<<(NN * OUTD + 255) / 256, 256, 0, stream>>>(hid1, lin2_w, lin2_b, out);
}
